// GNN_20109036879932
// MI455X (gfx1250) — matrix operand order repaired and checked by reading, not compiled, not run on hardware
//
#include <hip/hip_runtime.h>
#include <hip/hip_bf16.h>
#include <stddef.h>

// ---- sizes (fixed by the reference) ----
#define BATCH   128
#define NNODES  1024
#define NEDGES  2048
#define DEPTH   4

typedef __attribute__((ext_vector_type(16))) __bf16        v16bf;
typedef __attribute__((ext_vector_type(8)))  float         v8f;
typedef __attribute__((ext_vector_type(4)))  unsigned int  v4u;

union BF16x16 {
    v16bf          v;
    v4u            q[2];
    unsigned int   u[8];
    unsigned short s[16];
};

__device__ __forceinline__ unsigned short f2bf(float f) {
    unsigned int u = __float_as_uint(f);
    unsigned int r = u + 0x7FFFu + ((u >> 16) & 1u);   // round-to-nearest-even
    return (unsigned short)(r >> 16);
}

// Two f32 -> packed bf16 pair (round-to-nearest, ties up):
// one add per value + one v_perm_b32 to pack the two high halves.
__device__ __forceinline__ unsigned int pack_bf16_pair(float a, float b) {
    unsigned int ua = __float_as_uint(a) + 0x8000u;
    unsigned int ub = __float_as_uint(b) + 0x8000u;
    return __builtin_amdgcn_perm(ub, ua, 0x07060302u);  // {ub[31:16], ua[31:16]}
}

// -------------------------------------------------------------------------
// Setup kernels
// -------------------------------------------------------------------------

// A0 (f32 [N][E]) -> bf16 row-major and bf16 transposed [E][N]
__global__ void conv_a0_kernel(const float* __restrict__ A0,
                               unsigned short* __restrict__ A0b,
                               unsigned short* __restrict__ A0Tb) {
    int i = blockIdx.x * 256 + threadIdx.x;
    if (i < NNODES * NEDGES) {
        int n = i >> 11;            // / NEDGES
        int e = i & (NEDGES - 1);
        unsigned short b = f2bf(A0[i]);
        A0b[i]               = b;
        A0Tb[e * NNODES + n] = b;
    }
}

// m1vec[n] = sum_e A0[n,e]*q0[e] - D[n]
__global__ void m1_kernel(const float* __restrict__ A0,
                          const float* __restrict__ q0,
                          const float* __restrict__ D,
                          float* __restrict__ m1vec) {
    __shared__ float red[256];
    int n = blockIdx.x;
    float s = 0.f;
    for (int e = threadIdx.x; e < NEDGES; e += 256)
        s += A0[(size_t)n * NEDGES + e] * q0[e];
    red[threadIdx.x] = s;
    __syncthreads();
    for (int w = 128; w > 0; w >>= 1) {
        if (threadIdx.x < w) red[threadIdx.x] += red[threadIdx.x + w];
        __syncthreads();
    }
    if (threadIdx.x == 0) m1vec[n] = red[0] - D[n];
}

// mat1[e] = B_mat[e,:] . S_vec
__global__ void mat1_kernel(const float* __restrict__ Bm,
                            const float* __restrict__ Sv,
                            float* __restrict__ mat1) {
    int e = blockIdx.x * 256 + threadIdx.x;
    if (e < NEDGES) {
        float s = 0.f;
#pragma unroll
        for (int p = 0; p < 8; ++p) s += Bm[e * 8 + p] * Sv[p];
        mat1[e] = s;
    }
}

// Hc = H ; q = 0 ; m1buf[b,n] = m1vec[n]
__global__ void init_kernel(const float* __restrict__ H,
                            const float* __restrict__ m1vec,
                            float* __restrict__ Hc,
                            float* __restrict__ q,
                            float* __restrict__ m1buf) {
    int i = blockIdx.x * 256 + threadIdx.x;     // grid covers BATCH*NEDGES
    if (i < BATCH * NEDGES) q[i] = 0.f;
    if (i < BATCH * NNODES) {
        Hc[i]    = H[i];
        m1buf[i] = m1vec[i & (NNODES - 1)];
    }
}

// -------------------------------------------------------------------------
// Fused per-element MLP (in -> 64 -> 64 -> 1, relu), accumulated into `acc`.
// Middle 64x64 layer runs on v_wmma_f32_16x16x32_bf16.
// NIN = 2 (node MLP: x = [Hc, m1buf]) or 1 (edge MLP: x = m2buf).
// Also emits a bf16 mirror of the updated accumulator for the next GEMM.
// -------------------------------------------------------------------------
template <int NIN>
__global__ void __launch_bounds__(256, 1)
mlp_kernel(const float* x0buf,
           const float* x1buf,
           const float* __restrict__ W1,   // [NIN][64]
           const float* __restrict__ B1,   // [64]
           const float* __restrict__ W2,   // [64][64]
           const float* __restrict__ B2,   // [64]
           const float* __restrict__ W3,   // [64]
           const float* __restrict__ B3,   // [1]
           float* acc,                     // read+write, [M]
           unsigned short* __restrict__ accb, // bf16 mirror
           int M) {
    __shared__ float sW1[128], sB1[64], sW2[64 * 64], sB2[64], sW3[64], sB3;
    int tid = threadIdx.x;
    for (int i = tid; i < NIN * 64; i += 256) sW1[i] = W1[i];
    for (int i = tid; i < 64; i += 256) { sB1[i] = B1[i]; sB2[i] = B2[i]; sW3[i] = W3[i]; }
    for (int i = tid; i < 64 * 64; i += 256) sW2[i] = W2[i];
    if (tid == 0) sB3 = B3[0];
    __syncthreads();

    const int lane = tid & 31;
    const int l15  = lane & 15;
    const int hi   = lane >> 4;

    // B-operand (W2) register tiles: [ntile j][kstep s]
    // B layout: lane = column, k interleaved in 8-blocks per half-wave.
    BF16x16 breg[4][2];
#pragma unroll
    for (int j = 0; j < 4; ++j)
#pragma unroll
        for (int s = 0; s < 2; ++s)
#pragma unroll
            for (int i = 0; i < 16; i += 2) {
                int k = s * 32 + ((i < 8) ? (hi * 8 + i) : (16 + hi * 8 + (i - 8)));
                breg[j][s].u[i >> 1] = pack_bf16_pair(
                    sW2[k * 64 + j * 16 + l15],
                    sW2[(k + 1) * 64 + j * 16 + l15]);
            }

    const int wid = blockIdx.x * (blockDim.x >> 5) + (tid >> 5);
    const int nw  = gridDim.x * (blockDim.x >> 5);
    const int ntile = M >> 4;

    for (int t = wid; t < ntile; t += nw) {
        int r = t * 16 + l15;
        float x0 = x0buf[r];
        float x1 = (NIN == 2) ? x1buf[r] : 0.f;

        // Layer 1 in VALU, packed straight into WMMA A-operand layout:
        // lane = row, k interleaved in 8-blocks per half-wave.
        BF16x16 areg[2];
#pragma unroll
        for (int s = 0; s < 2; ++s)
#pragma unroll
            for (int i = 0; i < 16; i += 2) {
                int k0 = s * 32 + ((i < 8) ? (hi * 8 + i) : (16 + hi * 8 + (i - 8)));
                float h0 = sB1[k0] + x0 * sW1[k0];
                float h1 = sB1[k0 + 1] + x0 * sW1[k0 + 1];
                if (NIN == 2) {
                    h0 += x1 * sW1[64 + k0];
                    h1 += x1 * sW1[64 + k0 + 1];
                }
                areg[s].u[i >> 1] =
                    pack_bf16_pair(fmaxf(h0, 0.f), fmaxf(h1, 0.f));
            }

        // Layers 2+3 fused per N-tile: keep only ONE 8-VGPR accumulator live.
        float p[8];
#pragma unroll
        for (int v = 0; v < 8; ++v) p[v] = 0.f;
#pragma unroll
        for (int j = 0; j < 4; ++j) {
            float bb = sB2[j * 16 + l15];
            v8f c;
#pragma unroll
            for (int v = 0; v < 8; ++v) c[v] = bb;
            c = __builtin_amdgcn_wmma_f32_16x16x32_bf16(
                false, areg[0].v, false, breg[j][0].v, (short)0, c, false, false);
            c = __builtin_amdgcn_wmma_f32_16x16x32_bf16(
                false, areg[1].v, false, breg[j][1].v, (short)0, c, false, false);
            float w3 = sW3[j * 16 + l15];
#pragma unroll
            for (int v = 0; v < 8; ++v) p[v] += fmaxf(c[v], 0.f) * w3;
        }

        // Value-halving butterfly reduction over the 16 lanes of each half:
        // 8 values x 16 lanes -> 1 value per lane in 8 shuffles.
        // Final owner: lane l15 holds out[vidx], vidx = 4*b0 + 2*b1 + b2 of l15.
#pragma unroll
        for (int v = 0; v < 4; ++v) {                       // mask 1, r = 4
            float sent = (l15 & 1) ? p[v] : p[v + 4];
            float kept = (l15 & 1) ? p[v + 4] : p[v];
            p[v] = kept + __shfl_xor(sent, 1, 32);
        }
#pragma unroll
        for (int v = 0; v < 2; ++v) {                       // mask 2, r = 2
            float sent = (l15 & 2) ? p[v] : p[v + 2];
            float kept = (l15 & 2) ? p[v + 2] : p[v];
            p[v] = kept + __shfl_xor(sent, 2, 32);
        }
        {                                                   // mask 4, r = 1
            float sent = (l15 & 4) ? p[0] : p[1];
            float kept = (l15 & 4) ? p[1] : p[0];
            p[0] = kept + __shfl_xor(sent, 4, 32);
        }
        p[0] += __shfl_xor(p[0], 8, 32);                    // mask 8, plain

        if (l15 < 8) {
            int vidx = ((l15 & 1) << 2) | (l15 & 2) | ((l15 >> 2) & 1);
            int ro = t * 16 + hi * 8 + vidx;
            float val = acc[ro] + p[0] + sB3;
            acc[ro]  = val;
            accb[ro] = f2bf(val);
        }
    }
}

// -------------------------------------------------------------------------
// bf16 GEMM via WMMA:  out[b, col] = f(colvec[col], sum_k A[b,k]*Bm[col,k])
//   A  : [BATCH][K]  bf16 row-major  (Hc or q)
//   Bm : [Ncols][K]  bf16 (A0T for spmm1, A0 for spmm2) -> contiguous k loads
//   MODE 0: out = colvec - acc   (m2 = mat1 - A0^T Hc)
//   MODE 1: out = acc - colvec   (m1b = A0 q - D)
// -------------------------------------------------------------------------
template <int MODE>
__global__ void __launch_bounds__(256, 1)
spmm_kernel(const unsigned short* __restrict__ A,
            const unsigned short* __restrict__ Bm,
            const float* __restrict__ colvec,
            float* __restrict__ out,
            int N, int K) {
    int tid  = threadIdx.x;
    int lane = tid & 31, l15 = lane & 15, hi = lane >> 4;
    int wid  = blockIdx.x * (blockDim.x >> 5) + (tid >> 5);
    int nw   = gridDim.x * (blockDim.x >> 5);
    int ntiles = (N >> 4) * (BATCH >> 4);

    for (int t = wid; t < ntiles; t += nw) {
        int mt = t & ((BATCH >> 4) - 1);
        int nt = t >> 3;                     // BATCH/16 == 8
        const unsigned short* pa = A  + (size_t)(mt * 16 + l15) * K;
        const unsigned short* pb = Bm + (size_t)(nt * 16 + l15) * K;

        v8f c;
#pragma unroll
        for (int v = 0; v < 8; ++v) c[v] = 0.f;

#pragma unroll 4
        for (int kk = 0; kk < K; kk += 32) {
            BF16x16 a, b;
            int oa = kk + hi * 8;            // A: interleaved 8-blocks
            a.q[0] = *(const v4u*)(pa + oa);
            a.q[1] = *(const v4u*)(pa + oa + 16);
            int ob = kk + hi * 8;            // B: interleaved 8-blocks
            b.q[0] = *(const v4u*)(pb + ob);
            b.q[1] = *(const v4u*)(pb + ob + 16);
            c = __builtin_amdgcn_wmma_f32_16x16x32_bf16(
                false, a.v, false, b.v, (short)0, c, false, false);
        }

        int col  = nt * 16 + l15;
        float cv = colvec[col];
#pragma unroll
        for (int v = 0; v < 8; ++v) {
            int row = mt * 16 + hi * 8 + v;
            out[(size_t)row * N + col] = (MODE == 0) ? (cv - c[v]) : (c[v] - cv);
        }
    }
}

// -------------------------------------------------------------------------
// Launch
// -------------------------------------------------------------------------
extern "C" void kernel_launch(void* const* d_in, const int* in_sizes, int n_in,
                              void* d_out, int out_size, void* d_ws, size_t ws_size,
                              hipStream_t stream) {
    (void)in_sizes; (void)n_in; (void)out_size; (void)ws_size;

    const float* H   = (const float*)d_in[0];
    const float* A0  = (const float*)d_in[1];
    const float* D   = (const float*)d_in[2];
    const float* Bm  = (const float*)d_in[3];
    const float* Sv  = (const float*)d_in[4];
    const float* q0  = (const float*)d_in[5];
    const float* nW1 = (const float*)d_in[6];
    const float* nb1 = (const float*)d_in[7];
    const float* nW2 = (const float*)d_in[8];
    const float* nb2 = (const float*)d_in[9];
    const float* nW3 = (const float*)d_in[10];
    const float* nb3 = (const float*)d_in[11];
    const float* eW1 = (const float*)d_in[12];
    const float* eb1 = (const float*)d_in[13];
    const float* eW2 = (const float*)d_in[14];
    const float* eb2 = (const float*)d_in[15];
    const float* eW3 = (const float*)d_in[16];
    const float* eb3 = (const float*)d_in[17];

    float* Hc = (float*)d_out;                       // [BATCH*NNODES]
    float* q  = Hc + BATCH * NNODES;                 // [BATCH*NEDGES]

    char* ws = (char*)d_ws;
    unsigned short* A0b  = (unsigned short*)ws; ws += (size_t)NNODES * NEDGES * 2;
    unsigned short* A0Tb = (unsigned short*)ws; ws += (size_t)NNODES * NEDGES * 2;
    unsigned short* Hcb  = (unsigned short*)ws; ws += (size_t)BATCH * NNODES * 2;
    unsigned short* qb   = (unsigned short*)ws; ws += (size_t)BATCH * NEDGES * 2;
    float* m1buf = (float*)ws; ws += (size_t)BATCH * NNODES * 4;
    float* m2buf = (float*)ws; ws += (size_t)BATCH * NEDGES * 4;
    float* m1vec = (float*)ws; ws += (size_t)NNODES * 4;
    float* mat1  = (float*)ws; ws += (size_t)NEDGES * 4;

    // ---- setup ----
    conv_a0_kernel<<<(NNODES * NEDGES) / 256, 256, 0, stream>>>(A0, A0b, A0Tb);
    m1_kernel<<<NNODES, 256, 0, stream>>>(A0, q0, D, m1vec);
    mat1_kernel<<<NEDGES / 256, 256, 0, stream>>>(Bm, Sv, mat1);
    init_kernel<<<(BATCH * NEDGES) / 256, 256, 0, stream>>>(H, m1vec, Hc, q, m1buf);

    // ---- DEPTH unrolled iterations ----
    for (int it = 0; it < DEPTH; ++it) {
        // Hc += nodeMLP([Hc, m1buf]); Hcb = bf16(Hc)
        mlp_kernel<2><<<512, 256, 0, stream>>>(Hc, m1buf,
                                               nW1, nb1, nW2, nb2, nW3, nb3,
                                               Hc, Hcb, BATCH * NNODES);
        // m2 = mat1 - A0^T Hc
        spmm_kernel<0><<<128, 256, 0, stream>>>(Hcb, A0Tb, mat1, m2buf,
                                                NEDGES, NNODES);
        // q += edgeMLP(m2); qb = bf16(q)
        mlp_kernel<1><<<512, 256, 0, stream>>>(m2buf, nullptr,
                                               eW1, eb1, eW2, eb2, eW3, eb3,
                                               q, qb, BATCH * NEDGES);
        // m1buf = A0 q - D
        spmm_kernel<1><<<64, 256, 0, stream>>>(qb, A0b, D, m1buf,
                                               NNODES, NEDGES);
    }
}
